// CustomGNN_12068858102316
// MI455X (gfx1250) — matrix operand order repaired and checked by reading, not compiled, not run on hardware
//
#include <hip/hip_runtime.h>
#include <hip/hip_bf16.h>
#include <math.h>

#define IN_DIM 512
#define HID    256
#define OUTD   128

typedef __attribute__((ext_vector_type(16))) _Float16 v16h;
typedef __attribute__((ext_vector_type(8)))  float    v8f;

// ---------------------------------------------------------------------------
// A-fragment load: f32 row-major -> v16h in contiguous k order (k = 16*(lane/16) + s), same order as B.
// lane<16 : row=lane,    K = [0..15]   (of the 32-wide K step)
// lane>=16: row=lane-16, K = [16..31]
// Caller pre-offsets the row pointer by (lane>=16 ? 16 : 0).
// ---------------------------------------------------------------------------
__device__ __forceinline__ v16h load_a16_f32(const float* __restrict__ f) {
  float4 x0 = *(const float4*)(f);
  float4 x1 = *(const float4*)(f + 4);
  float4 x2 = *(const float4*)(f + 8);
  float4 x3 = *(const float4*)(f + 12);
  v16h a;
  a[0]  = (_Float16)x0.x; a[1]  = (_Float16)x0.y; a[2]  = (_Float16)x0.z; a[3]  = (_Float16)x0.w;
  a[4]  = (_Float16)x1.x; a[5]  = (_Float16)x1.y; a[6]  = (_Float16)x1.z; a[7]  = (_Float16)x1.w;
  a[8]  = (_Float16)x2.x; a[9]  = (_Float16)x2.y; a[10] = (_Float16)x2.z; a[11] = (_Float16)x2.w;
  a[12] = (_Float16)x3.x; a[13] = (_Float16)x3.y; a[14] = (_Float16)x3.z; a[15] = (_Float16)x3.w;
  return a;
}

// ---------------------------------------------------------------------------
// GEMM (+optional second product): out = act(A@B [+ A2@B2] + bias)
// A,A2: f32 [M,K] row-major.  Bt,B2t: f16 [Nout,K] (pre-transposed weights).
// One wave => 16x64 output strip (4 WMMA accumulators).
// ACT: 0 = relu, 1 = tanh
// ---------------------------------------------------------------------------
template<bool DUAL, int ACT>
__global__ void __launch_bounds__(128)
gemm_wmma_k(const float* __restrict__ A,  const _Float16* __restrict__ Bt,
            const float* __restrict__ A2, const _Float16* __restrict__ B2t,
            const float* __restrict__ bias, float* __restrict__ out,
            int M, int K, int Nout)
{
  const int wave       = blockIdx.x * (blockDim.x >> 5) + (threadIdx.x >> 5);
  const int tilesN     = Nout >> 6;                 // 64-wide N strips
  const int totalWaves = (M >> 4) * tilesN;
  if (wave >= totalWaves) return;                   // whole-wave uniform exit
  const int tileM = wave / tilesN;
  const int tileN = wave % tilesN;
  const int lane  = threadIdx.x & 31;
  const int lh    = lane & 15;
  const int up    = lane >> 4;

  const float* ap  = A + (size_t)(tileM * 16 + lh) * K + up * 16;
  const float* ap2 = DUAL ? (A2 + (size_t)(tileM * 16 + lh) * K + up * 16) : nullptr;

  // B fragment: lane<16 -> col=lane, K 0..15 of step; lane>=16 -> K 16..31
  const _Float16* bp[4];
  const _Float16* bp2[4];
#pragma unroll
  for (int j = 0; j < 4; ++j) {
    const int col = tileN * 64 + j * 16 + lh;
    bp[j] = Bt + (size_t)col * K + up * 16;
    if (DUAL) bp2[j] = B2t + (size_t)col * K + up * 16;
  }

  v8f acc[4] = {};

  for (int k0 = 0; k0 < K; k0 += 32) {
    v16h a = load_a16_f32(ap + k0);
#pragma unroll
    for (int j = 0; j < 4; ++j) {
      v16h b = *(const v16h*)(bp[j] + k0);
      acc[j] = __builtin_amdgcn_wmma_f32_16x16x32_f16(
                   false, a, false, b, (short)0, acc[j], false, false);
    }
    if (DUAL) {
      v16h a2 = load_a16_f32(ap2 + k0);
#pragma unroll
      for (int j = 0; j < 4; ++j) {
        v16h b2 = *(const v16h*)(bp2[j] + k0);
        acc[j] = __builtin_amdgcn_wmma_f32_16x16x32_f16(
                     false, a2, false, b2, (short)0, acc[j], false, false);
      }
    }
  }

  // C/D layout: lane<16 -> N=lane, M rows 0..7 in vgpr r; lane>=16 -> M rows 8..15
  const int mbase = tileM * 16 + up * 8;
#pragma unroll
  for (int j = 0; j < 4; ++j) {
    const int nn = tileN * 64 + j * 16 + lh;
    const float bv = bias[nn];
#pragma unroll
    for (int r = 0; r < 8; ++r) {
      float v = acc[j][r] + bv;
      if (ACT == 0) v = v > 0.f ? v : 0.f;
      else          v = tanhf(v);
      out[(size_t)(mbase + r) * Nout + nn] = v;
    }
  }
}

// ---------------------------------------------------------------------------
// Weight convert+transpose: W f32 [K,Nout] -> Wt f16 [Nout,K]
// ---------------------------------------------------------------------------
__global__ void wcvt_k(const float* __restrict__ W, _Float16* __restrict__ Wt,
                       int K, int Nout)
{
  int idx = blockIdx.x * blockDim.x + threadIdx.x;
  if (idx >= K * Nout) return;
  int k = idx / Nout;
  int n = idx - k * Nout;
  Wt[(size_t)n * K + k] = (_Float16)W[idx];
}

// ---------------------------------------------------------------------------
// Edge-phase kernels (per-dest per-channel softmax aggregation)
// ---------------------------------------------------------------------------
__global__ void fill3_k(float* __restrict__ mx, float* __restrict__ s,
                        float* __restrict__ num, int n)
{
  int i = blockIdx.x * blockDim.x + threadIdx.x;
  if (i >= n) return;
  mx[i]  = __int_as_float(0xFF800000);  // -inf
  s[i]   = 0.f;
  num[i] = 0.f;
}

__device__ __forceinline__ void atomicMaxF(float* addr, float v) {
  // IEEE ordering trick (NaN-free data)
  if (v >= 0.f) atomicMax((int*)addr, __float_as_int(v));
  else          atomicMin((unsigned int*)addr, __float_as_uint(v));
}

__global__ void edge_max_k(const float* __restrict__ x,
                           const int* __restrict__ src, const int* __restrict__ dst,
                           float* __restrict__ mx, int E)
{
  int idx = blockIdx.x * blockDim.x + threadIdx.x;
  if (idx >= E * 64) return;               // 64 chunks of 4 channels
  int e = idx >> 6;
  int c = (idx & 63) << 2;
  int s = src[e], d = dst[e];
  float4 m = *(const float4*)(x + (size_t)s * HID + c);
  float* mp = mx + (size_t)d * HID + c;
  atomicMaxF(mp + 0, m.x);
  atomicMaxF(mp + 1, m.y);
  atomicMaxF(mp + 2, m.z);
  atomicMaxF(mp + 3, m.w);
}

__global__ void edge_expsum_k(const float* __restrict__ x,
                              const int* __restrict__ src, const int* __restrict__ dst,
                              const float* __restrict__ mx,
                              float* __restrict__ ssum, float* __restrict__ num, int E)
{
  int idx = blockIdx.x * blockDim.x + threadIdx.x;
  if (idx >= E * 64) return;
  int e = idx >> 6;
  int c = (idx & 63) << 2;
  int s = src[e], d = dst[e];
  float4 m  = *(const float4*)(x  + (size_t)s * HID + c);
  float4 mv = *(const float4*)(mx + (size_t)d * HID + c);
  float4 w;
  w.x = __expf(m.x - mv.x); w.y = __expf(m.y - mv.y);
  w.z = __expf(m.z - mv.z); w.w = __expf(m.w - mv.w);
  float* sp = ssum + (size_t)d * HID + c;
  float* np = num  + (size_t)d * HID + c;
  atomicAdd(sp + 0, w.x);        atomicAdd(sp + 1, w.y);
  atomicAdd(sp + 2, w.z);        atomicAdd(sp + 3, w.w);
  atomicAdd(np + 0, w.x * m.x);  atomicAdd(np + 1, w.y * m.y);
  atomicAdd(np + 2, w.z * m.z);  atomicAdd(np + 3, w.w * m.w);
}

__global__ void normalize_k(float* __restrict__ num, const float* __restrict__ s, int n)
{
  int i = blockIdx.x * blockDim.x + threadIdx.x;
  if (i >= n) return;
  float sv = s[i];
  num[i] = (sv > 0.f) ? num[i] / sv : 0.f;   // isolated dest -> 0 (matches segment_sum)
}

// ---------------------------------------------------------------------------
// Orchestration
// ---------------------------------------------------------------------------
extern "C" void kernel_launch(void* const* d_in, const int* in_sizes, int n_in,
                              void* d_out, int out_size, void* d_ws, size_t ws_size,
                              hipStream_t stream)
{
  const int N = in_sizes[0] / IN_DIM;      // 50000
  const int E = in_sizes[1] / 2;           // 400000

  const float* feat = (const float*)d_in[0];
  const int*   src  = (const int*)d_in[1];
  const int*   dst  = src + E;
  const float* W1  = (const float*)d_in[2];  const float* b1  = (const float*)d_in[3];
  const float* W2  = (const float*)d_in[4];  const float* b2  = (const float*)d_in[5];
  const float* W3  = (const float*)d_in[6];  const float* b3  = (const float*)d_in[7];
  const float* Wl1 = (const float*)d_in[8];  const float* bl1 = (const float*)d_in[9];
  const float* Wr1 = (const float*)d_in[10];
  const float* WlH = (const float*)d_in[11]; const float* blH = (const float*)d_in[12];
  const float* WrH = (const float*)d_in[13];
  const float* Wl2 = (const float*)d_in[14]; const float* bl2 = (const float*)d_in[15];
  const float* Wr2 = (const float*)d_in[16];
  const float* Wp1 = (const float*)d_in[17]; const float* bp1 = (const float*)d_in[18];
  const float* Wp2 = (const float*)d_in[19]; const float* bp2 = (const float*)d_in[20];
  const float* Wp3 = (const float*)d_in[21]; const float* bp3 = (const float*)d_in[22];

  // workspace carve-up
  const size_t nh = (size_t)N * HID;
  float* buf0 = (float*)d_ws;
  float* buf1 = buf0 + nh;
  float* sbuf = buf1 + nh;
  float* numb = sbuf + nh;
  _Float16* wp   = (_Float16*)(numb + nh);
  _Float16* W1t  = wp;  wp += (size_t)HID * IN_DIM;
  _Float16* W2t  = wp;  wp += (size_t)HID * HID;
  _Float16* W3t  = wp;  wp += (size_t)HID * HID;
  _Float16* Wl1t = wp;  wp += (size_t)HID * HID;
  _Float16* Wr1t = wp;  wp += (size_t)HID * HID;
  _Float16* WlHt = wp;  wp += (size_t)HID * HID;
  _Float16* WrHt = wp;  wp += (size_t)HID * HID;
  _Float16* Wl2t = wp;  wp += (size_t)HID * HID;
  _Float16* Wr2t = wp;  wp += (size_t)HID * HID;
  _Float16* Wp1t = wp;  wp += (size_t)HID * HID;
  _Float16* Wp2t = wp;  wp += (size_t)HID * HID;
  _Float16* Wp3t = wp;  wp += (size_t)OUTD * HID;

  auto cvt = [&](const float* W, _Float16* Wt, int K, int Nn) {
    int t = K * Nn;
    wcvt_k<<<(t + 255) / 256, 256, 0, stream>>>(W, Wt, K, Nn);
  };
  cvt(W1,  W1t,  IN_DIM, HID);
  cvt(W2,  W2t,  HID, HID);   cvt(W3,  W3t,  HID, HID);
  cvt(Wl1, Wl1t, HID, HID);   cvt(Wr1, Wr1t, HID, HID);
  cvt(WlH, WlHt, HID, HID);   cvt(WrH, WrHt, HID, HID);
  cvt(Wl2, Wl2t, HID, HID);   cvt(Wr2, Wr2t, HID, HID);
  cvt(Wp1, Wp1t, HID, HID);   cvt(Wp2, Wp2t, HID, HID);
  cvt(Wp3, Wp3t, HID, OUTD);

  auto gemm = [&](const float* A, const _Float16* Bt, const float* bias,
                  float* out, int K, int Nout, int act) {
    int waves  = (N / 16) * (Nout / 64);
    int blocks = (waves + 3) / 4;
    if (act == 0)
      gemm_wmma_k<false, 0><<<blocks, 128, 0, stream>>>(A, Bt, nullptr, nullptr, bias, out, N, K, Nout);
    else
      gemm_wmma_k<false, 1><<<blocks, 128, 0, stream>>>(A, Bt, nullptr, nullptr, bias, out, N, K, Nout);
  };
  auto dual_gemm = [&](const float* A, const _Float16* Bt,
                       const float* A2, const _Float16* B2t,
                       const float* bias, float* out) {
    int waves  = (N / 16) * (HID / 64);
    int blocks = (waves + 3) / 4;
    gemm_wmma_k<true, 0><<<blocks, 128, 0, stream>>>(A, Bt, A2, B2t, bias, out, N, HID, HID);
  };

  const int nelems  = (int)nh;
  const int nblk    = (nelems + 255) / 256;
  const int eth     = E * 64;
  const int eblk    = (eth + 255) / 256;

  auto sage = [&](const float* xin, float* mxbuf,
                  const _Float16* Wlt, const _Float16* Wrt,
                  const float* bl, float* xout) {
    fill3_k<<<nblk, 256, 0, stream>>>(mxbuf, sbuf, numb, nelems);
    edge_max_k<<<eblk, 256, 0, stream>>>(xin, src, dst, mxbuf, E);
    edge_expsum_k<<<eblk, 256, 0, stream>>>(xin, src, dst, mxbuf, sbuf, numb, E);
    normalize_k<<<nblk, 256, 0, stream>>>(numb, sbuf, nelems);
    dual_gemm(numb, Wlt, xin, Wrt, bl, xout);   // xout may alias mxbuf (mx dead here)
  };

  // MLP encoder
  gemm(feat, W1t, b1, buf0, IN_DIM, HID, 0);
  gemm(buf0, W2t, b2, buf1, HID, HID, 0);
  gemm(buf1, W3t, b3, buf0, HID, HID, 0);
  // 3 x SAGE softmax conv (ping-pong; mx shares the output buffer)
  sage(buf0, buf1, Wl1t, Wr1t, bl1, buf1);
  sage(buf1, buf0, WlHt, WrHt, blH, buf0);
  sage(buf0, buf1, Wl2t, Wr2t, bl2, buf1);
  // MLP head
  gemm(buf1, Wp1t, bp1, buf0, HID, HID, 0);
  gemm(buf0, Wp2t, bp2, buf1, HID, HID, 0);
  gemm(buf1, Wp3t, bp3, (float*)d_out, HID, OUTD, 1);   // tanh
}
